// Layer_19035295056636
// MI455X (gfx1250) — matrix operand order repaired and checked by reading, not compiled, not run on hardware
//
#include <hip/hip_runtime.h>
#include <hip/hip_bf16.h>
#include <math.h>

typedef _Float16 f16;
typedef __attribute__((ext_vector_type(16))) _Float16 v16h;
typedef __attribute__((ext_vector_type(8)))  _Float16 v8h;
typedef __attribute__((ext_vector_type(8)))  float    v8f;

#define B_  2
#define S_  512
#define D_  256
#define FF_ 1024
#define H_  4
#define N_  512
#define P_  32
#define K_  16
#define KP_ 4
#define R_  64
#define DH_ 64
#define T_  (B_*S_)

// ---------------------------------------------------------------------------
// Batched WMMA GEMM, unit K-stride for both operands:
//   A(m,k) = A[m*asM + k]         (row-major activations)
//   B(k,n) = B[n*bsN + k]         (column-major / pre-transposed weights)
//   C[m,n] = alpha * sum_k A*B + bias[n]   (+= existing C, optional sigmoid,
//                                           optional f16 mirror)
// One wave computes a 16x32 tile: one A fragment feeds two
// v_wmma_f32_16x16x32_f16 ops. All fragment loads are aligned b128.
// ---------------------------------------------------------------------------
__global__ __launch_bounds__(256) void k_gemm(
    const f16* __restrict__ A, long asM, long aBy, long aBz,
    const f16* __restrict__ Bm, long bsN, long bBy, long bBz,
    const float* __restrict__ bias,
    float* __restrict__ C, long ldc, long cBy, long cBz,
    f16* __restrict__ C16, long ldc16, long c16By, long c16Bz,
    int M, int Nd, int Kd, float alpha, int accum, int act)
{
  const int wave = threadIdx.x >> 5;
  const int lane = threadIdx.x & 31;
  const int ntN  = Nd >> 5;                      // 32-wide column tiles
  const int tiles = (M >> 4) * ntN;
  const int tile  = blockIdx.x * (blockDim.x >> 5) + wave;
  if (tile >= tiles) return;                     // wave-uniform: EXEC stays ~0
  const int tm = tile / ntN, tn = tile - tm * ntN;
  const f16* Ab = A  + (long)blockIdx.y * aBy + (long)blockIdx.z * aBz;
  const f16* Bb = Bm + (long)blockIdx.y * bBy + (long)blockIdx.z * bBz;
  const int M0 = tm << 4, N0 = tn << 5;
  const int l15 = lane & 15;
  const long arow  = (long)(M0 + l15) * asM;     // A row this lane supplies
  const long bcol0 = (long)(N0 + l15) * bsN;     // B col, left 16x16 tile
  const long bcol1 = (long)(N0 + 16 + l15) * bsN;// B col, right 16x16 tile
  const int kA = (lane & 16) ? 8  : 0;           // ISA 16-bit A lane layout
  const int kB = (lane & 16) ? 8  : 0;           // ISA 16-bit B lane layout

  v8f acc0 = {}, acc1 = {};
  for (int k0 = 0; k0 < Kd; k0 += 32) {
    union { v16h v; v8h h[2]; } a, b0, b1;
    a.h[0]  = *(const v8h*)(Ab + arow  + k0 + kA);        // K = kA..kA+7
    a.h[1]  = *(const v8h*)(Ab + arow  + k0 + kA + 16);   // K = kA+16..kA+23
    b0.h[0] = *(const v8h*)(Bb + bcol0 + k0 + kB);        // K = kB..kB+7
    b0.h[1] = *(const v8h*)(Bb + bcol0 + k0 + kB + 16);   // K = kB+16..kB+23
    b1.h[0] = *(const v8h*)(Bb + bcol1 + k0 + kB);
    b1.h[1] = *(const v8h*)(Bb + bcol1 + k0 + kB + 16);
    if (k0 + 64 < Kd) {                          // prefetch 2 K-steps ahead
      __builtin_prefetch(Ab + arow  + k0 + 64 + kA, 0, 0);
      __builtin_prefetch(Bb + bcol0 + k0 + 64 + kB, 0, 0);
      __builtin_prefetch(Bb + bcol1 + k0 + 64 + kB, 0, 0);
    }
    acc0 = __builtin_amdgcn_wmma_f32_16x16x32_f16(false, a.v, false, b0.v,
                                                  (short)0, acc0, false, false);
    acc1 = __builtin_amdgcn_wmma_f32_16x16x32_f16(false, a.v, false, b1.v,
                                                  (short)0, acc1, false, false);
  }

  const int rbase = M0 + ((lane >> 4) << 3);     // VGPR r -> row (+8 hi lanes)
  const int nc0 = N0 + l15, nc1 = N0 + 16 + l15;
  const float bb0 = bias ? bias[nc0] : 0.0f;
  const float bb1 = bias ? bias[nc1] : 0.0f;
  const long cb   = (long)blockIdx.y * cBy   + (long)blockIdx.z * cBz;
  const long cb16 = (long)blockIdx.y * c16By + (long)blockIdx.z * c16Bz;
#pragma unroll
  for (int r = 0; r < 8; ++r) {
    const int row = rbase + r;
    float v0 = acc0[r] * alpha + bb0;
    float v1 = acc1[r] * alpha + bb1;
    if (C) {
      const long i0 = (long)row * ldc + nc0 + cb;
      const long i1 = (long)row * ldc + nc1 + cb;
      if (accum) { v0 += C[i0]; v1 += C[i1]; }
      if (act == 1) {
        v0 = 1.0f / (1.0f + __expf(-v0));
        v1 = 1.0f / (1.0f + __expf(-v1));
      }
      C[i0] = v0; C[i1] = v1;
    } else if (act == 1) {
      v0 = 1.0f / (1.0f + __expf(-v0));
      v1 = 1.0f / (1.0f + __expf(-v1));
    }
    if (C16) {
      C16[(long)row * ldc16 + nc0 + cb16] = (f16)v0;
      C16[(long)row * ldc16 + nc1 + cb16] = (f16)v1;
    }
  }
}

// ---------------------------------------------------------------------------
// LayerNorm: one block per token, both norm1 and norm2 from shared mu/var.
// ---------------------------------------------------------------------------
__global__ __launch_bounds__(256) void k_ln(const float* __restrict__ x,
    const float* __restrict__ g1, const float* __restrict__ b1,
    const float* __restrict__ g2, const float* __restrict__ b2,
    float* __restrict__ n1f, f16* __restrict__ n1h, float* __restrict__ n2f)
{
  const int t = blockIdx.x, d = threadIdx.x;
  __shared__ float red[256];
  const float v = x[(long)t * D_ + d];
  red[d] = v; __syncthreads();
  for (int s = 128; s > 0; s >>= 1) { if (d < s) red[d] += red[d + s]; __syncthreads(); }
  const float mu = red[0] * (1.0f / D_);
  __syncthreads();
  const float c = v - mu;
  red[d] = c * c; __syncthreads();
  for (int s = 128; s > 0; s >>= 1) { if (d < s) red[d] += red[d + s]; __syncthreads(); }
  const float rs = rsqrtf(red[0] * (1.0f / D_) + 1e-5f);
  const float a1 = c * rs * g1[d] + b1[d];
  const float a2 = c * rs * g2[d] + b2[d];
  n1f[(long)t * D_ + d] = a1;
  n1h[(long)t * D_ + d] = (f16)a1;
  n2f[(long)t * D_ + d] = a2;
}

__global__ __launch_bounds__(256) void k_f32_to_f16(const float* __restrict__ s,
                                                    f16* __restrict__ d, int n)
{
  for (int i = blockIdx.x * 256 + threadIdx.x; i < n; i += gridDim.x * 256)
    d[i] = (f16)s[i];
}

// f32 [R][C] -> f16 transposed [C][R] (weights become unit-K-stride B operands)
__global__ __launch_bounds__(256) void k_convT(const float* __restrict__ s,
                                               f16* __restrict__ d, int Rr, int Cc)
{
  const int n = Rr * Cc;
  for (int i = blockIdx.x * 256 + threadIdx.x; i < n; i += gridDim.x * 256) {
    const int r = i / Cc, c = i - r * Cc;
    d[(long)c * Rr + r] = (f16)s[i];
  }
}

// vh [b*S+s][h*64+dh] -> vt [b][h*64+dh][s]  (unit-K-stride B for attn@V)
__global__ __launch_bounds__(256) void k_vt(const f16* __restrict__ vh,
                                            f16* __restrict__ vt)
{
  const int i = blockIdx.x * 256 + threadIdx.x;
  if (i < T_ * D_) {
    const int t = i >> 8, c = i & (D_ - 1);
    const int b = t >> 9, s = t & (S_ - 1);
    vt[(long)b * D_ * S_ + (long)c * S_ + s] = vh[i];
  }
}

// Softmax over last dim (512) of attention scores; emits f16 probs.
__global__ __launch_bounds__(256) void k_softmax_att(const float* __restrict__ att,
                                                     f16* __restrict__ attnh)
{
  const long row = blockIdx.x;
  const int tid = threadIdx.x;
  __shared__ float red[256];
  const float* p = att + row * S_;
  const float v0 = p[tid], v1 = p[tid + 256];
  red[tid] = fmaxf(v0, v1); __syncthreads();
  for (int s = 128; s > 0; s >>= 1) { if (tid < s) red[tid] = fmaxf(red[tid], red[tid + s]); __syncthreads(); }
  const float m = red[0];
  __syncthreads();
  const float e0 = __expf(v0 - m), e1 = __expf(v1 - m);
  red[tid] = e0 + e1; __syncthreads();
  for (int s = 128; s > 0; s >>= 1) { if (tid < s) red[tid] += red[tid + s]; __syncthreads(); }
  const float inv = 1.0f / red[0];
  attnh[row * S_ + tid]       = (f16)(e0 * inv);
  attnh[row * S_ + tid + 256] = (f16)(e1 * inv);
}

// Path gating: w = softmax([n1,ctx] @ Wpath + bpath); fuse final scores in place.
__global__ __launch_bounds__(256) void k_path(const float* __restrict__ n1f,
    const float* __restrict__ ctxf, const float* __restrict__ Wpath,
    const float* __restrict__ bpath, float* __restrict__ tok,
    const float* __restrict__ ctxsc)
{
  const int t = blockIdx.x, tid = threadIdx.x;
  __shared__ float r0[256], r1[256];
  const float a = n1f[(long)t * D_ + tid];
  const float c = ctxf[(long)t * D_ + tid];
  r0[tid] = a * Wpath[tid * 2 + 0] + c * Wpath[(D_ + tid) * 2 + 0];
  r1[tid] = a * Wpath[tid * 2 + 1] + c * Wpath[(D_ + tid) * 2 + 1];
  __syncthreads();
  for (int s = 128; s > 0; s >>= 1) {
    if (tid < s) { r0[tid] += r0[tid + s]; r1[tid] += r1[tid + s]; }
    __syncthreads();
  }
  const float l0 = r0[0] + bpath[0], l1 = r1[0] + bpath[1];
  const float m = fmaxf(l0, l1);
  const float e0 = __expf(l0 - m), e1 = __expf(l1 - m);
  const float inv = 1.0f / (e0 + e1);
  const float w0 = e0 * inv, w1 = e1 * inv;
  for (int n = tid; n < N_; n += 256)
    tok[(long)t * N_ + n] = w0 * tok[(long)t * N_ + n] + w1 * ctxsc[(long)t * N_ + n];
}

// Top-16 of 512 per token (iterative argmax, ties -> lowest index, descending),
// then softmax over the 16 scores. Writes indices to ws and to d_out tail.
__global__ __launch_bounds__(256) void k_topk(const float* __restrict__ scores,
    float* __restrict__ topw, int* __restrict__ topidx, int* __restrict__ outIdx)
{
  const int t = blockIdx.x, tid = threadIdx.x;
  __shared__ float sv[N_];
  __shared__ float rv[256];
  __shared__ int   ri[256];
  __shared__ float svals[K_];
  __shared__ int   chosen[K_];
  sv[tid]       = scores[(long)t * N_ + tid];
  sv[tid + 256] = scores[(long)t * N_ + tid + 256];
  __syncthreads();
  for (int it = 0; it < K_; ++it) {
    const float v0 = sv[tid], v1 = sv[tid + 256];
    float mv; int mi;
    if (v1 > v0) { mv = v1; mi = tid + 256; } else { mv = v0; mi = tid; }
    rv[tid] = mv; ri[tid] = mi; __syncthreads();
    for (int s = 128; s > 0; s >>= 1) {
      if (tid < s) {
        if (rv[tid + s] > rv[tid] ||
            (rv[tid + s] == rv[tid] && ri[tid + s] < ri[tid])) {
          rv[tid] = rv[tid + s]; ri[tid] = ri[tid + s];
        }
      }
      __syncthreads();
    }
    if (tid == 0) { svals[it] = rv[0]; chosen[it] = ri[0]; sv[ri[0]] = -1e30f; }
    __syncthreads();
  }
  if (tid == 0) {
    const float m = svals[0];
    float e[K_]; float sum = 0.0f;
    for (int j = 0; j < K_; ++j) { e[j] = __expf(svals[j] - m); sum += e[j]; }
    const float inv = 1.0f / sum;
    for (int j = 0; j < K_; ++j) {
      topw[t * K_ + j]   = e[j] * inv;
      topidx[t * K_ + j] = chosen[j];
      outIdx[t * K_ + j] = chosen[j];
    }
  }
}

// Per-token neuron interaction: 4-head 16x16 attention over selected neurons,
// gated aggregation, combined = n2 + aggregated.
__global__ __launch_bounds__(256) void k_interact(const int* __restrict__ idx,
    const float* __restrict__ topw, const float* __restrict__ nq,
    const float* __restrict__ nk, const float* __restrict__ nv,
    const float* __restrict__ neurons, const float* __restrict__ n2f,
    float* __restrict__ aggf, f16* __restrict__ aggh,
    float* __restrict__ combf, f16* __restrict__ combh)
{
  const int t = blockIdx.x, tid = threadIdx.x;
  __shared__ int   sidx[K_];
  __shared__ float sw[K_];
  __shared__ float aw[H_][K_][K_];
  __shared__ float gates[K_][D_];
  if (tid < K_) { sidx[tid] = idx[t * K_ + tid]; sw[tid] = topw[t * K_ + tid]; }
  __syncthreads();
  const int q = tid >> 4, kk = tid & 15;          // 256 threads = 16x16 (q,k)
  for (int h = 0; h < H_; ++h) {
    const float* qp = nq + (long)sidx[q]  * D_ + h * DH_;
    const float* kp = nk + (long)sidx[kk] * D_ + h * DH_;
    float s = 0.0f;
    for (int d = 0; d < DH_; ++d) s += qp[d] * kp[d];
    aw[h][q][kk] = s * 0.125f;                    // 1/sqrt(64)
  }
  __syncthreads();
  if (tid < H_ * K_) {                            // softmax over k per (h,q)
    const int h = tid >> 4, qq = tid & 15;
    float m = -1e30f;
    for (int j = 0; j < K_; ++j) m = fmaxf(m, aw[h][qq][j]);
    float sum = 0.0f;
    for (int j = 0; j < K_; ++j) { const float e = __expf(aw[h][qq][j] - m); aw[h][qq][j] = e; sum += e; }
    const float inv = 1.0f / sum;
    for (int j = 0; j < K_; ++j) aw[h][qq][j] *= inv;
  }
  __syncthreads();
  for (int e = tid; e < K_ * D_; e += 256) {      // gates[q][d]
    const int qq = e >> 8, d = e & 255, h = d >> 6;
    float s = 0.0f;
    for (int j = 0; j < K_; ++j)
      s += aw[h][qq][j] * nv[(long)sidx[j] * D_ + d];   // nv already sigmoided
    gates[qq][d] = s;
  }
  __syncthreads();
  const int d = tid;
  float s = 0.0f;
  for (int j = 0; j < K_; ++j)
    s += sw[j] * neurons[(long)sidx[j] * D_ + d] * gates[j][d];
  aggf[(long)t * D_ + d] = s;
  aggh[(long)t * D_ + d] = (f16)s;
  const float cb = n2f[(long)t * D_ + d] + s;
  combf[(long)t * D_ + d] = cb;
  combh[(long)t * D_ + d] = (f16)cb;
}

// Pattern top-4 of 32 + gathered LoRA: h_pattern[f] = sum_p tpw[p]*(comb@A[p])@B[p]
__global__ __launch_bounds__(256) void k_lora(const float* __restrict__ ps,
    const float* __restrict__ combf, const float* __restrict__ upA,
    const float* __restrict__ upB, float* __restrict__ hpat)
{
  const int t = blockIdx.x, tid = threadIdx.x;
  __shared__ float sps[P_];
  __shared__ int   spi[KP_];
  __shared__ float spw[KP_];
  __shared__ float hmid[KP_][R_];
  __shared__ float scomb[D_];
  scomb[tid] = combf[(long)t * D_ + tid];
  if (tid < P_) sps[tid] = ps[(long)t * P_ + tid];
  __syncthreads();
  if (tid == 0) {
    float vals[KP_];
    for (int it = 0; it < KP_; ++it) {
      float best = -1e30f; int bi = 0;
      for (int n = 0; n < P_; ++n)
        if (sps[n] > best) { best = sps[n]; bi = n; }
      spi[it] = bi; vals[it] = best; sps[bi] = -1e30f;
    }
    const float m = vals[0];
    float e[KP_]; float sum = 0.0f;
    for (int j = 0; j < KP_; ++j) { e[j] = __expf(vals[j] - m); sum += e[j]; }
    for (int j = 0; j < KP_; ++j) spw[j] = e[j] / sum;
  }
  __syncthreads();
  {                                               // hmid[p][r], 256 threads
    const int p = tid >> 6, r = tid & 63;
    const float* Ap = upA + (long)spi[p] * D_ * R_ + r;
    float s = 0.0f;
    for (int d = 0; d < D_; ++d) s += scomb[d] * Ap[(long)d * R_];
    hmid[p][r] = s;
  }
  __syncthreads();
  for (int f = tid; f < FF_; f += 256) {
    float s = 0.0f;
    for (int p = 0; p < KP_; ++p) {
      const float* Bp = upB + (long)spi[p] * R_ * FF_ + f;
      float acc = 0.0f;
      for (int r = 0; r < R_; ++r) acc += hmid[p][r] * Bp[(long)r * FF_];
      s += spw[p] * acc;
    }
    hpat[(long)t * FF_ + f] = s;
  }
}

__global__ __launch_bounds__(256) void k_gelu(const float* __restrict__ ub,
    const float* __restrict__ hpat, f16* __restrict__ hh, int n)
{
  const int i = blockIdx.x * 256 + threadIdx.x;
  if (i < n) {
    const float hx = 0.1f * ub[i] + 0.9f * hpat[i];
    const float g = 0.5f * hx * (1.0f + erff(hx * 0.70710678118654752f));
    hh[i] = (f16)g;
  }
}

__global__ __launch_bounds__(256) void k_resid(const float* __restrict__ x,
    const float* __restrict__ ff, float* __restrict__ out, int n)
{
  const int i = blockIdx.x * 256 + threadIdx.x;
  if (i < n) out[i] = x[i] + ff[i];
}

// ---------------------------------------------------------------------------
extern "C" void kernel_launch(void* const* d_in, const int* in_sizes, int n_in,
                              void* d_out, int out_size, void* d_ws, size_t ws_size,
                              hipStream_t stream)
{
  (void)in_sizes; (void)n_in; (void)out_size; (void)ws_size;
  const float* x        = (const float*)d_in[0];
  const float* g1       = (const float*)d_in[1];
  const float* b1       = (const float*)d_in[2];
  const float* g2       = (const float*)d_in[3];
  const float* b2       = (const float*)d_in[4];
  const float* Wq       = (const float*)d_in[5];
  const float* bq       = (const float*)d_in[6];
  const float* Wk       = (const float*)d_in[7];
  const float* bk       = (const float*)d_in[8];
  const float* Wv       = (const float*)d_in[9];
  const float* bv       = (const float*)d_in[10];
  const float* Wpath    = (const float*)d_in[11];
  const float* bpath    = (const float*)d_in[12];
  const float* neurons  = (const float*)d_in[13];
  const float* Wnq      = (const float*)d_in[14];
  const float* bnq      = (const float*)d_in[15];
  const float* Wnk      = (const float*)d_in[16];
  const float* bnk      = (const float*)d_in[17];
  const float* Wnv      = (const float*)d_in[18];
  const float* bnv      = (const float*)d_in[19];
  const float* pq       = (const float*)d_in[20];
  const float* upA      = (const float*)d_in[21];
  const float* upB      = (const float*)d_in[22];
  const float* Wub      = (const float*)d_in[23];
  const float* bub      = (const float*)d_in[24];
  const float* Wd       = (const float*)d_in[25];
  const float* bd       = (const float*)d_in[26];

  // --- workspace bump allocator (deterministic layout each call) ---
  char* wp = (char*)d_ws;
  auto alloc = [&](size_t bytes) -> char* {
    char* r = wp; wp += (bytes + 255) & ~(size_t)255; return r;
  };
  f16*  n1h    = (f16*)alloc(sizeof(f16) * T_ * D_);
  f16*  WqT    = (f16*)alloc(sizeof(f16) * D_ * D_);   // [dout][din]
  f16*  WkT    = (f16*)alloc(sizeof(f16) * D_ * D_);
  f16*  WvT    = (f16*)alloc(sizeof(f16) * D_ * D_);
  f16*  WnqT   = (f16*)alloc(sizeof(f16) * D_ * D_);
  f16*  WnkT   = (f16*)alloc(sizeof(f16) * D_ * D_);
  f16*  WnvT   = (f16*)alloc(sizeof(f16) * D_ * D_);
  f16*  neur16 = (f16*)alloc(sizeof(f16) * N_ * D_);   // [n][d] (already B-layout)
  f16*  pq16   = (f16*)alloc(sizeof(f16) * P_ * D_);   // [p][d]
  f16*  WubT   = (f16*)alloc(sizeof(f16) * D_ * FF_);  // [ff][d]
  f16*  WdT    = (f16*)alloc(sizeof(f16) * FF_ * D_);  // [d][ff]
  f16*  qh     = (f16*)alloc(sizeof(f16) * T_ * D_);
  f16*  kh     = (f16*)alloc(sizeof(f16) * T_ * D_);
  f16*  vh     = (f16*)alloc(sizeof(f16) * T_ * D_);
  f16*  vt     = (f16*)alloc(sizeof(f16) * T_ * D_);   // [b][h*64+dh][s]
  f16*  attnh  = (f16*)alloc(sizeof(f16) * (size_t)B_ * H_ * S_ * S_);
  f16*  ctxh   = (f16*)alloc(sizeof(f16) * T_ * D_);
  f16*  aggh   = (f16*)alloc(sizeof(f16) * T_ * D_);
  f16*  combh  = (f16*)alloc(sizeof(f16) * T_ * D_);
  f16*  hh     = (f16*)alloc(sizeof(f16) * T_ * FF_);
  float* n1f   = (float*)alloc(sizeof(float) * T_ * D_);
  float* n2f   = (float*)alloc(sizeof(float) * T_ * D_);
  float* attf  = (float*)alloc(sizeof(float) * (size_t)B_ * H_ * S_ * S_);
  float* ctxf  = (float*)alloc(sizeof(float) * T_ * D_);
  float* tokf  = (float*)alloc(sizeof(float) * T_ * N_);
  float* ctxsc = (float*)alloc(sizeof(float) * T_ * N_);
  float* nqf   = (float*)alloc(sizeof(float) * N_ * D_);
  float* nkf   = (float*)alloc(sizeof(float) * N_ * D_);
  float* nvf   = (float*)alloc(sizeof(float) * N_ * D_);
  float* topw  = (float*)alloc(sizeof(float) * T_ * K_);
  float* aggf  = (float*)alloc(sizeof(float) * T_ * D_);
  float* combf = (float*)alloc(sizeof(float) * T_ * D_);
  float* psf   = (float*)alloc(sizeof(float) * T_ * P_);
  float* hpatf = (float*)alloc(sizeof(float) * T_ * FF_);
  float* ubf   = (float*)alloc(sizeof(float) * T_ * FF_);
  float* fff   = (float*)alloc(sizeof(float) * T_ * D_);
  int*   topidx = (int*)alloc(sizeof(int) * T_ * K_);
  int*   outIdx = ((int*)d_out) + T_ * D_;   // tuple output #2: topk_idx

  auto conv = [&](const float* s, f16* d, int n) {
    k_f32_to_f16<<<(n + 255) / 256, 256, 0, stream>>>(s, d, n);
  };
  auto convT = [&](const float* s, f16* d, int Rr, int Cc) {
    k_convT<<<(Rr * Cc + 255) / 256, 256, 0, stream>>>(s, d, Rr, Cc);
  };
  auto gemm = [&](const f16* A, long asM, long aBy, long aBz,
                  const f16* Bmat, long bsN, long bBy, long bBz,
                  const float* bias,
                  float* C, long ldc, long cBy, long cBz,
                  f16* C16, long ldc16, long c16By, long c16Bz,
                  int M, int Nd, int Kd, float alpha, int accum, int act,
                  int nby, int nbz) {
    const int tiles = (M >> 4) * (Nd >> 5);
    dim3 grid((tiles + 7) / 8, nby, nbz);
    k_gemm<<<grid, 256, 0, stream>>>(A, asM, aBy, aBz,
                                     Bmat, bsN, bBy, bBz, bias,
                                     C, ldc, cBy, cBz, C16, ldc16, c16By, c16Bz,
                                     M, Nd, Kd, alpha, accum, act);
  };

  // 1) LayerNorms
  k_ln<<<T_, 256, 0, stream>>>(x, g1, b1, g2, b2, n1f, n1h, n2f);

  // 2) weights -> f16 (transposed where needed so B is unit-K-stride)
  convT(Wq, WqT, D_, D_);   convT(Wk, WkT, D_, D_);   convT(Wv, WvT, D_, D_);
  convT(Wnq, WnqT, D_, D_); convT(Wnk, WnkT, D_, D_); convT(Wnv, WnvT, D_, D_);
  convT(Wub, WubT, D_, FF_); convT(Wd, WdT, FF_, D_);
  conv(neurons, neur16, N_ * D_); conv(pq, pq16, P_ * D_);

  // 3) q/k/v = n1 @ W + b   (f16 outputs, layout [T, h*64+dh])
  gemm(n1h, D_, 0, 0, WqT, D_, 0, 0, bq, nullptr, 0, 0, 0,
       qh, D_, 0, 0, T_, D_, D_, 1.0f, 0, 0, 1, 1);
  gemm(n1h, D_, 0, 0, WkT, D_, 0, 0, bk, nullptr, 0, 0, 0,
       kh, D_, 0, 0, T_, D_, D_, 1.0f, 0, 0, 1, 1);
  gemm(n1h, D_, 0, 0, WvT, D_, 0, 0, bv, nullptr, 0, 0, 0,
       vh, D_, 0, 0, T_, D_, D_, 1.0f, 0, 0, 1, 1);
  k_vt<<<(T_ * D_ + 255) / 256, 256, 0, stream>>>(vh, vt);

  // 4) attention scores[b,h,q,k] = q.kT / 8  (batch y=B, z=H)
  //    B(k=dh, n=kpos) = kh[b*S*D + kpos*D + h*64 + dh]  -> bsN=D, unit K
  gemm(qh, D_, (long)S_ * D_, DH_,
       kh, D_, (long)S_ * D_, DH_, nullptr,
       attf, S_, (long)H_ * S_ * S_, (long)S_ * S_,
       nullptr, 0, 0, 0,
       S_, S_, DH_, 0.125f, 0, 0, B_, H_);

  // 5) row softmax -> f16 probs
  k_softmax_att<<<B_ * H_ * S_, 256, 0, stream>>>(attf, attnh);

  // 6) context[b,q,h,dh] = attn @ v
  //    B(k=kpos, n=dh) = vt[b*D*S + (h*64+dh)*S + kpos]  -> bsN=S, unit K
  gemm(attnh, S_, (long)H_ * S_ * S_, (long)S_ * S_,
       vt, S_, (long)D_ * S_, (long)DH_ * S_, nullptr,
       ctxf, D_, (long)S_ * D_, DH_,
       ctxh, D_, (long)S_ * D_, DH_,
       S_, DH_, S_, 1.0f, 0, 0, B_, H_);

  // 7) neuron scores (token & context), B(k=d, n=neuron) = neurons[n*D+d]
  gemm(n1h, D_, 0, 0, neur16, D_, 0, 0, nullptr,
       tokf, N_, 0, 0, nullptr, 0, 0, 0, T_, N_, D_, 1.0f, 0, 0, 1, 1);
  gemm(ctxh, D_, 0, 0, neur16, D_, 0, 0, nullptr,
       ctxsc, N_, 0, 0, nullptr, 0, 0, 0, T_, N_, D_, 1.0f, 0, 0, 1, 1);

  // 8) path gating mixes scores in place; 9) top-16 + softmax weights
  k_path<<<T_, 256, 0, stream>>>(n1f, ctxf, Wpath, bpath, tokf, ctxsc);
  k_topk<<<T_, 256, 0, stream>>>(tokf, topw, topidx, outIdx);

  // 10) neuron q/k/v once for all 512 neurons (selected@W == gather of these)
  gemm(neur16, D_, 0, 0, WnqT, D_, 0, 0, bnq,
       nqf, D_, 0, 0, nullptr, 0, 0, 0, N_, D_, D_, 1.0f, 0, 0, 1, 1);
  gemm(neur16, D_, 0, 0, WnkT, D_, 0, 0, bnk,
       nkf, D_, 0, 0, nullptr, 0, 0, 0, N_, D_, D_, 1.0f, 0, 0, 1, 1);
  gemm(neur16, D_, 0, 0, WnvT, D_, 0, 0, bnv,
       nvf, D_, 0, 0, nullptr, 0, 0, 0, N_, D_, D_, 1.0f, 0, 1, 1, 1); // sigmoid

  // 11) per-token neuron interaction, aggregation, combined = n2 + agg
  k_interact<<<T_, 256, 0, stream>>>(topidx, topw, nqf, nkf, nvf, neurons, n2f,
                                     aggf, aggh, combf, combh);

  // 12) pattern scores = 0.5*(agg@pqT)/16 + 0.5*(ctx@pqT)
  gemm(aggh, D_, 0, 0, pq16, D_, 0, 0, nullptr,
       psf, P_, 0, 0, nullptr, 0, 0, 0, T_, P_, D_, 0.5f / 16.0f, 0, 0, 1, 1);
  gemm(ctxh, D_, 0, 0, pq16, D_, 0, 0, nullptr,
       psf, P_, 0, 0, nullptr, 0, 0, 0, T_, P_, D_, 0.5f, 1, 0, 1, 1);

  // 13) pattern top-4 + LoRA
  k_lora<<<T_, 256, 0, stream>>>(psf, combf, upA, upB, hpatf);

  // 14) ub = combined @ Wub + bub ; 15) gelu(0.1*ub + 0.9*h_pattern) -> f16
  gemm(combh, D_, 0, 0, WubT, D_, 0, 0, bub,
       ubf, FF_, 0, 0, nullptr, 0, 0, 0, T_, FF_, D_, 1.0f, 0, 0, 1, 1);
  k_gelu<<<(T_ * FF_ + 255) / 256, 256, 0, stream>>>(ubf, hpatf, hh, T_ * FF_);

  // 16) out = h @ Wd + bd ; 17) residual
  gemm(hh, FF_, 0, 0, WdT, FF_, 0, 0, bd,
       fff, D_, 0, 0, nullptr, 0, 0, 0, T_, D_, FF_, 1.0f, 0, 0, 1, 1);
  k_resid<<<(T_ * D_ + 255) / 256, 256, 0, stream>>>(x, fff, (float*)d_out, T_ * D_);
}
